// SASelfAtt_2138893713812
// MI455X (gfx1250) — compile-verified, matrix operand order checked by reading, not run on hardware
//
#include <hip/hip_runtime.h>
#include <hip/hip_bf16.h>

#define DD    576      // feature dim
#define NROWS 512      // b*c = 4*128
#define KSTEPS 18      // 576 / 32

typedef __attribute__((ext_vector_type(16))) __bf16         v16bf;
typedef __attribute__((ext_vector_type(8)))  float          v8f;
typedef __attribute__((ext_vector_type(8)))  unsigned short u16x8;
typedef __attribute__((ext_vector_type(16))) unsigned short u16x16;
typedef __attribute__((ext_vector_type(4)))  int            v4i;

// Address-space-qualified pointer types for the async-to-LDS builtin
// (param 1 is 'v4i __device__*' per clang's diagnostic; param 2 is the LDS side).
typedef __attribute__((address_space(1))) v4i* as1_v4i_p;
typedef __attribute__((address_space(3))) v4i* as3_v4i_p;

#if defined(__has_builtin)
#  if __has_builtin(__builtin_amdgcn_global_load_async_to_lds_b128)
#    define HAVE_ASYNC_LDS 1
#  endif
#endif

__device__ __forceinline__ unsigned short f32_to_bf16_rne(float f) {
    unsigned u = __float_as_uint(f);
    unsigned lsb = (u >> 16) & 1u;
    u += 0x7fffu + lsb;                 // round-to-nearest-even
    return (unsigned short)(u >> 16);
}

// Load a 16-half WMMA operand fragment: halves[0..7] at p, halves[8..15] at p+16
__device__ __forceinline__ v16bf load_frag(const unsigned short* p) {
    u16x8 lo = *(const u16x8*)(p);
    u16x8 hi = *(const u16x8*)(p + 16);
    u16x16 f;
#pragma unroll
    for (int i = 0; i < 8; ++i) { f[i] = lo[i]; f[i + 8] = hi[i]; }
    return __builtin_bit_cast(v16bf, f);
}

// ---------------------------------------------------------------------------
// Kernel 1: convert x (512x576) and Wq/Wk/Wv (576x576 each) fp32 -> bf16
// ---------------------------------------------------------------------------
__global__ void cvt_bf16_kernel(const float* __restrict__ x,
                                const float* __restrict__ Wq,
                                const float* __restrict__ Wk,
                                const float* __restrict__ Wv,
                                unsigned short* __restrict__ xb,
                                unsigned short* __restrict__ wb) {
    int idx = blockIdx.x * 256 + threadIdx.x;
    const int NX = NROWS * DD;   // 294912
    const int NW = DD * DD;      // 331776
    if (idx < NX) { xb[idx] = f32_to_bf16_rne(x[idx]); return; }
    idx -= NX;
    if (idx < NW) { wb[idx] = f32_to_bf16_rne(Wq[idx]); return; }
    idx -= NW;
    if (idx < NW) { wb[NW + idx] = f32_to_bf16_rne(Wk[idx]); return; }
    idx -= NW;
    if (idx < NW) { wb[2 * NW + idx] = f32_to_bf16_rne(Wv[idx]); }
}

// ---------------------------------------------------------------------------
// Kernel 2: fused Q/K/V GEMM via v_wmma_f32_16x16x32_bf16.
// grid = (9, 8, 3): blockIdx.x = group of 4 N-tiles, blockIdx.y*4+wave = M-tile,
// blockIdx.z = matrix (Q/K/V). Each wave computes a 16x64 output strip:
// one A fragment per k-step is reused across 4 WMMAs (register blocking).
//   A lane layout: lane&15 = row, halves[0..7] at k=kb+8*(lane>>4), [8..15] at +16
//   B lane layout: lane&15 = out col n; B[k,n]=W[n,k] -> contiguous row of W
//   C/D layout:    lane&15 = col, VGPR r = row r (lanes 0-15) / row 8+r (16-31)
// ---------------------------------------------------------------------------
__global__ void qkv_wmma_kernel(const unsigned short* __restrict__ xb,
                                const unsigned short* __restrict__ wb,
                                const float* __restrict__ bq,
                                const float* __restrict__ bk,
                                const float* __restrict__ bv,
                                float* __restrict__ qkv) {
    int wave  = threadIdx.x >> 5;
    int lane  = threadIdx.x & 31;
    int half_ = lane >> 4;       // 0 or 1
    int l16   = lane & 15;

    int mat = blockIdx.z;                 // 0..2
    int tm  = (blockIdx.y << 2) + wave;   // 0..31
    int tg  = blockIdx.x;                 // 0..8 -> cols [tg*64, tg*64+64)

    const unsigned short* A  = xb + (size_t)((tm << 4) + l16) * DD;
    const unsigned short* Wm = wb + (size_t)mat * DD * DD;
    const unsigned short* B0 = Wm + (size_t)((tg << 6) +  0 + l16) * DD;
    const unsigned short* B1 = Wm + (size_t)((tg << 6) + 16 + l16) * DD;
    const unsigned short* B2 = Wm + (size_t)((tg << 6) + 32 + l16) * DD;
    const unsigned short* B3 = Wm + (size_t)((tg << 6) + 48 + l16) * DD;

    v8f c0 = {}, c1 = {}, c2 = {}, c3 = {};
#pragma unroll
    for (int kk = 0; kk < KSTEPS; ++kk) {
        int k0 = (kk << 5) + (half_ << 3);
        v16bf av = load_frag(A + k0);
        v16bf b0 = load_frag(B0 + k0);
        v16bf b1 = load_frag(B1 + k0);
        v16bf b2 = load_frag(B2 + k0);
        v16bf b3 = load_frag(B3 + k0);
        // (neg_a, A, neg_b, B, c_mod, C, reuse_a, reuse_b)
        c0 = __builtin_amdgcn_wmma_f32_16x16x32_bf16(false, av, false, b0, (short)0, c0, false, false);
        c1 = __builtin_amdgcn_wmma_f32_16x16x32_bf16(false, av, false, b1, (short)0, c1, false, false);
        c2 = __builtin_amdgcn_wmma_f32_16x16x32_bf16(false, av, false, b2, (short)0, c2, false, false);
        c3 = __builtin_amdgcn_wmma_f32_16x16x32_bf16(false, av, false, b3, (short)0, c3, false, false);
    }

    const float* bias = (mat == 0) ? bq : (mat == 1) ? bk : bv;
    float* out = qkv + (size_t)mat * NROWS * DD;
    int rowbase = (tm << 4) + (half_ << 3);

    v8f acc[4] = { c0, c1, c2, c3 };
#pragma unroll
    for (int t = 0; t < 4; ++t) {
        int col = (tg << 6) + (t << 4) + l16;
        float badd = bias[col];
#pragma unroll
        for (int r = 0; r < 8; ++r) {
            out[(size_t)(rowbase + r) * DD + col] = acc[t][r] + badd;
        }
    }
}

// ---------------------------------------------------------------------------
// Kernel 3: rank-1 attention. Block per row n; K[n,:],V[n,:] staged in LDS
// (via ASYNCcnt-tracked global_load_async_to_lds_b128 when available).
// Att[n,i] = sum_j e^{qK[j]-m} V[j] / sum_j e^{qK[j]-m},  q=Q[n,i],
// m = q>=0 ? q*max(K) : q*min(K)   (exact row-max of the rank-1 scores)
// ---------------------------------------------------------------------------
__global__ void attn_kernel(const float* __restrict__ qkv,
                            float* __restrict__ out) {
    const float* Q = qkv;
    const float* K = qkv + (size_t)NROWS * DD;
    const float* V = qkv + (size_t)2 * NROWS * DD;
    int n   = blockIdx.x;
    int tid = threadIdx.x;

    __shared__ float Ksh[DD];
    __shared__ float Vsh[DD];
    __shared__ float red[256];

#if defined(HAVE_ASYNC_LDS)
    // 576 floats = 144 lanes x 16B per row; lanes 0..143 issue one b128 each.
    // Generic->AS1: identical 64-bit value (flat addressing).
    // Generic->AS3: low 32 bits of the generic address ARE the LDS offset
    // (ISA 10.2).
    if (tid < DD / 4) {
        const float* gk = K + (size_t)n * DD + tid * 4;
        const float* gv = V + (size_t)n * DD + tid * 4;
        __builtin_amdgcn_global_load_async_to_lds_b128(
            (as1_v4i_p)(uintptr_t)gk,
            (as3_v4i_p)(unsigned)(uintptr_t)&Ksh[tid * 4], 0, 0);
        __builtin_amdgcn_global_load_async_to_lds_b128(
            (as1_v4i_p)(uintptr_t)gv,
            (as3_v4i_p)(unsigned)(uintptr_t)&Vsh[tid * 4], 0, 0);
    }
#  if __has_builtin(__builtin_amdgcn_s_wait_asynccnt)
    __builtin_amdgcn_s_wait_asynccnt(0);
#  else
    asm volatile("s_wait_asynccnt 0x0" ::: "memory");
#  endif
#else
    for (int j = tid; j < DD; j += 256) {
        Ksh[j] = K[(size_t)n * DD + j];
        Vsh[j] = V[(size_t)n * DD + j];
    }
#endif
    __syncthreads();

    float lmax = -__builtin_inff(), lmin = __builtin_inff();
    for (int j = tid; j < DD; j += 256) {
        float v = Ksh[j];
        lmax = fmaxf(lmax, v);
        lmin = fminf(lmin, v);
    }
    red[tid] = lmax; __syncthreads();
    for (int s = 128; s > 0; s >>= 1) {
        if (tid < s) red[tid] = fmaxf(red[tid], red[tid + s]);
        __syncthreads();
    }
    float kmax = red[0]; __syncthreads();
    red[tid] = lmin; __syncthreads();
    for (int s = 128; s > 0; s >>= 1) {
        if (tid < s) red[tid] = fminf(red[tid], red[tid + s]);
        __syncthreads();
    }
    float kmin = red[0];

    for (int i = tid; i < DD; i += 256) {
        float q = Q[(size_t)n * DD + i];
        float m = (q >= 0.f) ? q * kmax : q * kmin;
        float s = 0.f, t = 0.f;
        for (int j = 0; j < DD; ++j) {
            float e = __expf(__fmaf_rn(q, Ksh[j], -m));  // v_exp_f32
            s += e;
            t = __fmaf_rn(e, Vsh[j], t);
        }
        out[(size_t)n * DD + i] = t / s;
    }
}

// ---------------------------------------------------------------------------
extern "C" void kernel_launch(void* const* d_in, const int* in_sizes, int n_in,
                              void* d_out, int out_size, void* d_ws, size_t ws_size,
                              hipStream_t stream) {
    const float* x  = (const float*)d_in[0];
    const float* Wq = (const float*)d_in[1];
    const float* bq = (const float*)d_in[2];
    const float* Wk = (const float*)d_in[3];
    const float* bk = (const float*)d_in[4];
    const float* Wv = (const float*)d_in[5];
    const float* bv = (const float*)d_in[6];
    float* out = (float*)d_out;

    // Workspace layout (total ~5.9 MB):
    //   [0)                xb: 512*576 bf16
    //   [589824)           wb: 3 * 576*576 bf16
    //   [2580480)          qkv: 3 * 512*576 fp32 (Q, K, V)
    unsigned short* xb  = (unsigned short*)d_ws;
    unsigned short* wb  = xb + (size_t)NROWS * DD;
    float*          qkv = (float*)(wb + (size_t)3 * DD * DD);

    // 1) fp32 -> bf16 conversion: 294912 + 3*331776 = 1290240 elements
    cvt_bf16_kernel<<<5040, 256, 0, stream>>>(x, Wq, Wk, Wv, xb, wb);

    // 2) fused QKV GEMM: grid (Ngroups=9, Mgroups=8, mat=3), 4 waves/block,
    //    each wave -> 16x64 output strip, 72 WMMAs
    qkv_wmma_kernel<<<dim3(9, 8, 3), 128, 0, stream>>>(xb, wb, bq, bk, bv, qkv);

    // 3) rank-1 softmax attention: one block per row n
    attn_kernel<<<512, 256, 0, stream>>>(qkv, out);
}
